// Model_86011015070455
// MI455X (gfx1250) — compile-verified, matrix operand order checked by reading, not run on hardware
//
#include <hip/hip_runtime.h>

// ---- model constants --------------------------------------------------------
#define B_   128
#define N_   32
#define L_   100
#define U_   1024
#define XSTR 416   // per-agent K (400) padded to multiple of 32
#define K1T  416   // traj layer-1 padded K
#define K1I  832   // interaction layer-1 padded K (= 2 * XSTR)

typedef __attribute__((ext_vector_type(16))) __bf16 v16bf;
typedef __attribute__((ext_vector_type(8)))  float  v8f;
typedef __attribute__((ext_vector_type(4)))  unsigned int u32x4;

union FragU { v16bf v; u32x4 q[2]; };

__device__ __forceinline__ v8f bwmma(v16bf a, v16bf b, v8f c) {
    return __builtin_amdgcn_wmma_f32_16x16x32_bf16(
        /*neg_a=*/false, a, /*neg_b=*/false, b,
        /*c_mod=*/(short)0, c, /*reuse_a=*/false, /*reuse_b=*/false);
}

// ---- prep: x [B][4N][L] f32 -> xb [B*N][XSTR] bf16 (channel-major, zero pad)
__global__ void pack_x(const float* __restrict__ x, __bf16* __restrict__ xb) {
    int idx = blockIdx.x * blockDim.x + threadIdx.x;
    const int total = B_ * N_ * XSTR;
    if (idx >= total) return;
    int row = idx / XSTR;
    int k   = idx - row * XSTR;
    float v = 0.0f;
    if (k < 4 * L_) {
        int c = k / L_;
        int l = k - c * L_;
        int b = row >> 5;          // / N_
        int n = row & (N_ - 1);
        v = x[((size_t)(b * 4 * N_) + (n * 4 + c)) * L_ + l];
    }
    xb[idx] = (__bf16)v;
}

// ---- prep: w [Kin][O] f32 -> wt [O][Kp] bf16, segment-padded transpose
__global__ void conv_wT(const float* __restrict__ w, __bf16* __restrict__ out,
                        int O, int Kp, int segKp, int segKin) {
    int idx = blockIdx.x * blockDim.x + threadIdx.x;
    if (idx >= O * Kp) return;
    int o  = idx / Kp;
    int k  = idx - o * Kp;
    int seg = k / segKp;
    int kk  = k - seg * segKp;
    float v = 0.0f;
    if (kk < segKin) v = w[(size_t)(seg * segKin + kk) * O + o];
    out[idx] = (__bf16)v;
}

// ---- fused 3-layer MLP energy kernel ---------------------------------------
// Block: 512 threads (16 waves). Block tile: 32 rows x 1024 cols.
// Wave w owns cols [w*64, w*64+64) as 4 WMMA n-tiles, for 2 row tiles.
// K-loops fully unrolled with ping-pong fragment buffers so loads for step
// s+1 overlap the WMMAs of step s (partial loadcnt waits instead of full).
template <bool IS_PAIR>
__global__ __launch_bounds__(512, 1)
void mlp_energy(const __bf16* __restrict__ xb,
                const __bf16* __restrict__ w1t, const float* __restrict__ b1,
                const __bf16* __restrict__ w2t, const float* __restrict__ b2,
                const float* __restrict__ w3,  const float* __restrict__ b3,
                const float* __restrict__ edges,
                float* __restrict__ out) {
    constexpr int K1 = IS_PAIR ? K1I : K1T;
    constexpr int S1 = K1 / 32;
    constexpr int S2 = U_ / 32;
    constexpr int NT = 4;            // n-tiles per wave

    __shared__ __bf16 H1[32 * U_];   // 64 KB: layer-1 activations (bf16)
    __shared__ float  rowE[32];

    const int lane    = threadIdx.x & 31;
    const int wave    = threadIdx.x >> 5;   // 0..15
    const int lh      = lane >> 4;          // which K-half this lane loads
    const int ln      = lane & 15;          // row (A) / col (B) within tile
    const int colBase = wave * 64;

    if (threadIdx.x < 32) rowE[threadIdx.x] = 0.0f;

    // Per-row source pointers (pair rows gathered on the fly from xb)
    const __bf16* xi[2];
    const __bf16* xj[2];
#pragma unroll
    for (int rt = 0; rt < 2; ++rt) {
        int gr = blockIdx.x * 32 + rt * 16 + ln;
        if (IS_PAIR) {
            int b  = gr >> 10;            // / (N*N)
            int ij = gr & 1023;
            int i  = ij >> 5, j = ij & 31;
            xi[rt] = xb + (size_t)(b * N_ + i) * XSTR;
            xj[rt] = xb + (size_t)(b * N_ + j) * XSTR;
        } else {
            xi[rt] = xb + (size_t)gr * XSTR;
            xj[rt] = xi[rt];
        }
    }

    // A-fragment loader, layer 1. 416 is 32-aligned, so a whole 32-K step
    // lives entirely in xi or xj -> select by s (constexpr under unroll).
    auto loadA1 = [&](int s, FragU* a) {
        const bool hi  = IS_PAIR && (s * 32 >= XSTR);
        const int base = s * 32 - (hi ? XSTR : 0) + lh * 8;
#pragma unroll
        for (int rt = 0; rt < 2; ++rt) {
            const __bf16* p = (hi ? xj[rt] : xi[rt]) + base;
            a[rt].q[0] = *(const u32x4*)p;
            a[rt].q[1] = *(const u32x4*)(p + 16);
        }
    };
    auto loadB = [&](const __bf16* __restrict__ wt, int K, int s, FragU* bm) {
        const int c0 = s * 32 + lh * 8;
#pragma unroll
        for (int t = 0; t < NT; ++t) {
            const __bf16* wp = wt + (size_t)(colBase + t * 16 + ln) * K + c0;
            bm[t].q[0] = *(const u32x4*)wp;
            bm[t].q[1] = *(const u32x4*)(wp + 16);
        }
    };

    // ---------------- layer 1: H1 = relu(A @ W1 + b1) -----------------------
    v8f acc[2][NT];
#pragma unroll
    for (int rt = 0; rt < 2; ++rt)
#pragma unroll
        for (int t = 0; t < NT; ++t)
#pragma unroll
            for (int e = 0; e < 8; ++e) acc[rt][t][e] = 0.0f;

    {
        FragU aB[2][2], bB[2][NT];
        loadA1(0, aB[0]);
        loadB(w1t, K1, 0, bB[0]);
#pragma unroll
        for (int s = 0; s < S1; ++s) {
            const int cur = s & 1, nxt = cur ^ 1;
            if (s + 1 < S1) {
                loadA1(s + 1, aB[nxt]);
                loadB(w1t, K1, s + 1, bB[nxt]);
            }
#pragma unroll
            for (int t = 0; t < NT; ++t) {
                acc[0][t] = bwmma(aB[cur][0].v, bB[cur][t].v, acc[0][t]);
                acc[1][t] = bwmma(aB[cur][1].v, bB[cur][t].v, acc[1][t]);
            }
        }
    }

    // epilogue 1: bias + relu -> LDS (bf16)
#pragma unroll
    for (int t = 0; t < NT; ++t) {
        int col   = colBase + t * 16 + ln;
        float bia = b1[col];
#pragma unroll
        for (int rt = 0; rt < 2; ++rt)
#pragma unroll
            for (int jj = 0; jj < 8; ++jj) {
                int row = rt * 16 + lh * 8 + jj;
                float v = acc[rt][t][jj] + bia;
                v = v > 0.0f ? v : 0.0f;
                H1[row * U_ + col] = (__bf16)v;
            }
    }
    __syncthreads();

    // ---------------- layer 2: H2 = relu(H1 @ W2 + b2) ----------------------
    auto loadA2 = [&](int s, FragU* a) {
        const int c0 = s * 32 + lh * 8;
#pragma unroll
        for (int rt = 0; rt < 2; ++rt) {
            const __bf16* hp = &H1[(rt * 16 + ln) * U_ + c0];
            a[rt].q[0] = *(const u32x4*)hp;
            a[rt].q[1] = *(const u32x4*)(hp + 16);
        }
    };

    v8f acc2[2][NT];
#pragma unroll
    for (int rt = 0; rt < 2; ++rt)
#pragma unroll
        for (int t = 0; t < NT; ++t)
#pragma unroll
            for (int e = 0; e < 8; ++e) acc2[rt][t][e] = 0.0f;

    {
        FragU aB[2][2], bB[2][NT];
        loadA2(0, aB[0]);
        loadB(w2t, U_, 0, bB[0]);
#pragma unroll
        for (int s = 0; s < S2; ++s) {
            const int cur = s & 1, nxt = cur ^ 1;
            if (s + 1 < S2) {
                loadA2(s + 1, aB[nxt]);
                loadB(w2t, U_, s + 1, bB[nxt]);
            }
#pragma unroll
            for (int t = 0; t < NT; ++t) {
                acc2[0][t] = bwmma(aB[cur][0].v, bB[cur][t].v, acc2[0][t]);
                acc2[1][t] = bwmma(aB[cur][1].v, bB[cur][t].v, acc2[1][t]);
            }
        }
    }

    // epilogue 2 + layer 3: per-row energy = sum_col relu(h2) * w3[col] + b3
    float part[2][8];
#pragma unroll
    for (int rt = 0; rt < 2; ++rt)
#pragma unroll
        for (int jj = 0; jj < 8; ++jj) part[rt][jj] = 0.0f;

#pragma unroll
    for (int t = 0; t < NT; ++t) {
        int col   = colBase + t * 16 + ln;
        float bia = b2[col];
        float wv  = w3[col];
#pragma unroll
        for (int rt = 0; rt < 2; ++rt)
#pragma unroll
            for (int jj = 0; jj < 8; ++jj) {
                float h = acc2[rt][t][jj] + bia;
                h = h > 0.0f ? h : 0.0f;
                part[rt][jj] += h * wv;
            }
    }
#pragma unroll
    for (int rt = 0; rt < 2; ++rt)
#pragma unroll
        for (int jj = 0; jj < 8; ++jj)
            atomicAdd(&rowE[rt * 16 + lh * 8 + jj], part[rt][jj]);

    __syncthreads();

    if (threadIdx.x < 32) {
        int gr  = blockIdx.x * 32 + threadIdx.x;
        float e = rowE[threadIdx.x] + b3[0];
        int b;
        float val;
        if (IS_PAIR) {
            b   = gr >> 10;           // edges is flat [B*N*N] in the same order as gr
            val = edges[gr] * e;
        } else {
            b   = gr >> 5;
            val = e;
        }
        atomicAdd(&out[b], val);
    }
}

// ---- launcher ---------------------------------------------------------------
extern "C" void kernel_launch(void* const* d_in, const int* in_sizes, int n_in,
                              void* d_out, int out_size, void* d_ws, size_t ws_size,
                              hipStream_t stream) {
    (void)in_sizes; (void)n_in; (void)out_size; (void)ws_size;
    const float* x       = (const float*)d_in[0];
    const float* edges   = (const float*)d_in[1];
    const float* traj1_w = (const float*)d_in[2];
    const float* traj1_b = (const float*)d_in[3];
    const float* traj2_w = (const float*)d_in[4];
    const float* traj2_b = (const float*)d_in[5];
    const float* traj3_w = (const float*)d_in[6];
    const float* traj3_b = (const float*)d_in[7];
    const float* int1_w  = (const float*)d_in[8];
    const float* int1_b  = (const float*)d_in[9];
    const float* int2_w  = (const float*)d_in[10];
    const float* int2_b  = (const float*)d_in[11];
    const float* int3_w  = (const float*)d_in[12];
    const float* int3_b  = (const float*)d_in[13];
    float* out = (float*)d_out;

    // workspace layout (all sizes are multiples of 256 bytes)
    char* ws = (char*)d_ws;
    const size_t sz_xb  = (size_t)B_ * N_ * XSTR * 2;   // 3,407,872
    const size_t sz_w1t = (size_t)U_ * K1T * 2;          //   851,968
    const size_t sz_w2  = (size_t)U_ * U_ * 2;           // 2,097,152
    const size_t sz_w1i = (size_t)U_ * K1I * 2;          // 1,703,936
    size_t o = 0;
    __bf16* xb    = (__bf16*)(ws + o); o += sz_xb;
    __bf16* w1t_t = (__bf16*)(ws + o); o += sz_w1t;
    __bf16* w2t_t = (__bf16*)(ws + o); o += sz_w2;
    __bf16* w1t_i = (__bf16*)(ws + o); o += sz_w1i;
    __bf16* w2t_i = (__bf16*)(ws + o); o += sz_w2;

    // zero output accumulators
    hipMemsetAsync(d_out, 0, (size_t)B_ * sizeof(float), stream);

    // prep kernels
    {
        int total = B_ * N_ * XSTR;
        pack_x<<<(total + 255) / 256, 256, 0, stream>>>(x, xb);
    }
    {
        int total = U_ * K1T;
        conv_wT<<<(total + 255) / 256, 256, 0, stream>>>(traj1_w, w1t_t, U_, K1T, K1T, 4 * L_);
    }
    {
        int total = U_ * U_;
        conv_wT<<<(total + 255) / 256, 256, 0, stream>>>(traj2_w, w2t_t, U_, U_, U_, U_);
        conv_wT<<<(total + 255) / 256, 256, 0, stream>>>(int2_w,  w2t_i, U_, U_, U_, U_);
    }
    {
        int total = U_ * K1I;
        conv_wT<<<(total + 255) / 256, 256, 0, stream>>>(int1_w, w1t_i, U_, K1I, XSTR, 4 * L_);
    }

    // trajectory branch: 4096 rows -> 128 blocks of 32 rows
    mlp_energy<false><<<(B_ * N_) / 32, 512, 0, stream>>>(
        xb, w1t_t, traj1_b, w2t_t, traj2_b, traj3_w, traj3_b, nullptr, out);

    // interaction branch: 131072 rows -> 4096 blocks of 32 rows
    mlp_energy<true><<<(B_ * N_ * N_) / 32, 512, 0, stream>>>(
        xb, w1t_i, int1_b, w2t_i, int2_b, int3_w, int3_b, edges, out);
}
